// AttnDecoderRNN_65386582114504
// MI455X (gfx1250) — matrix operand order repaired and checked by reading, not compiled, not run on hardware
//
#include <hip/hip_runtime.h>
#include <hip/hip_bf16.h>

typedef __attribute__((ext_vector_type(16))) _Float16 v16h;
typedef __attribute__((ext_vector_type(8)))  float    v8f;
typedef __attribute__((ext_vector_type(4)))  unsigned int v4u;
typedef __attribute__((ext_vector_type(8)))  int      v8i;
typedef __attribute__((ext_vector_type(4)))  int      v4i;

#define H 1024
#define V 50257
#define L 64
#define B 512

// ---------------------------------------------------------------------------
// WMMA fragment helpers (wave32, 16x16x32 f16 -> f32), contiguous k order (k = 16*half + elem) in A and B.
// ---------------------------------------------------------------------------

static __device__ __forceinline__ v8f wmma_f16(v16h a, v16h b, v8f c) {
  // 8 args: (neg_a, A, neg_b, B, c_mod, C, reuse_a, reuse_b)
  return __builtin_amdgcn_wmma_f32_16x16x32_f16(false, a, false, b, (short)0, c,
                                                false, false);
}

// A fragment: 16x32 tile of row-major fp32 matrix, converted to f16.
// lanes 0-15 (row m0+lane): elems 0..15 = K 0..15
// lanes 16-31 (same rows):  elems 0..15 = K 16..31
static __device__ __forceinline__ v16h load_a_rm(const float* __restrict__ A,
                                                 int lda, int m0, int k0,
                                                 int lane) {
  const int half = lane >> 4;
  const int m = m0 + (lane & 15);
  const float4* p = (const float4*)(A + (long)m * lda + k0 + half * 16);
  float4 q0 = p[0], q1 = p[1];  // K = half*16 + 0..7
  float4 q2 = p[2], q3 = p[3];  // K = half*16 + 8..15
  v16h a;
  a[0] = (_Float16)q0.x; a[1] = (_Float16)q0.y; a[2] = (_Float16)q0.z; a[3] = (_Float16)q0.w;
  a[4] = (_Float16)q1.x; a[5] = (_Float16)q1.y; a[6] = (_Float16)q1.z; a[7] = (_Float16)q1.w;
  a[8]  = (_Float16)q2.x; a[9]  = (_Float16)q2.y; a[10] = (_Float16)q2.z; a[11] = (_Float16)q2.w;
  a[12] = (_Float16)q3.x; a[13] = (_Float16)q3.y; a[14] = (_Float16)q3.z; a[15] = (_Float16)q3.w;
  return a;
}

// B fragment: 32x16 tile of W stored row-major as (K_total x N_total), ldb = N.
// lanes 0-15 (col n0+lane): elem i = W[k0+i, n]        (K 0..15)
// lanes 16-31 (same cols):  elem i = W[k0+16+i, n]     (K 16..31)
static __device__ __forceinline__ v16h load_b_kn(const float* __restrict__ W,
                                                 long ldb, int k0, int n0,
                                                 int lane, int nmax) {
  const int half = lane >> 4;
  int n = n0 + (lane & 15);
  n = (n < nmax) ? n : (nmax - 1);
  const float* p = W + (long)(k0 + half * 16) * ldb + n;
  v16h b;
#pragma unroll
  for (int i = 0; i < 16; ++i) b[i] = (_Float16)p[(long)i * ldb];
  return b;
}

// B fragment from an LDS-staged 32xK-chunk tile with 65-dword padded rows
// (TDM pad_interval=64dw, pad_amount=1dw) -> conflict-free bank walk.
static __device__ __forceinline__ v16h load_b_lds(const float* __restrict__ s,
                                                  int n_local, int lane) {
  const int half = lane >> 4;
  const float* p = s + (half * 16) * 65 + n_local + (lane & 15);
  v16h b;
#pragma unroll
  for (int i = 0; i < 16; ++i) b[i] = (_Float16)p[i * 65];
  return b;
}

// B fragment for a transposed weight: logical B[k,n] = Wt[n, k], Wt row-major
// (N_total x K_total), ldk = K_total. Per-lane contiguous -> float4 loads.
static __device__ __forceinline__ v16h load_b_nk(const float* __restrict__ Wt,
                                                 int ldk, int k0, int n0,
                                                 int lane) {
  const int half = lane >> 4;
  const int n = n0 + (lane & 15);
  const float4* p = (const float4*)(Wt + (long)n * ldk + k0 + half * 16);
  float4 q0 = p[0], q1 = p[1], q2 = p[2], q3 = p[3];
  v16h b;
  b[0] = (_Float16)q0.x; b[1] = (_Float16)q0.y; b[2] = (_Float16)q0.z; b[3] = (_Float16)q0.w;
  b[4] = (_Float16)q1.x; b[5] = (_Float16)q1.y; b[6] = (_Float16)q1.z; b[7] = (_Float16)q1.w;
  b[8]  = (_Float16)q2.x; b[9]  = (_Float16)q2.y; b[10] = (_Float16)q2.z; b[11] = (_Float16)q2.w;
  b[12] = (_Float16)q3.x; b[13] = (_Float16)q3.y; b[14] = (_Float16)q3.z; b[15] = (_Float16)q3.w;
  return b;
}

// ---------------------------------------------------------------------------
// TDM: issue TENSOR_LOAD_TO_LDS of a 2D fp32 tile (32 rows x 64 cols) at
// `gsrc` (row stride V floats) into LDS offset `lds_off`, padded rows (65dw).
// D# layout per ISA 8.3/8.4; groups 2/3 zero (2D tensor). Wave-uniform args.
// This toolchain's builtin is the 6-arg form:
//   (uint32x4 g0, int32x8 g1, int32x4 g2, int32x4 g3, int32x8, i32 cpol)
// ---------------------------------------------------------------------------
static __device__ __forceinline__ void tdm_load_tile(const float* gsrc,
                                                     unsigned lds_off,
                                                     unsigned rem_cols) {
  const unsigned long long ga = (unsigned long long)gsrc;
  v4u g0;
  g0[0] = 1u;                                   // count=1, user descriptor
  g0[1] = lds_off;                              // lds_addr (bytes)
  g0[2] = (unsigned)(ga & 0xffffffffu);         // global_addr[31:0]
  g0[3] = (unsigned)(ga >> 32) | (2u << 30);    // global_addr[56:32] | type=2
  v8i g1;
  g1[0] = (2 << 16)                             // data_size = 4B
        | (1 << 20)                             // pad_enable
        | (5 << 22);                            // pad_interval: 64 dwords
                                                // pad_amount = 0 -> 1 dword
  g1[1] = (int)((rem_cols & 0xffffu) << 16);                      // tdim0 lo
  g1[2] = (int)((rem_cols >> 16) | (((unsigned)H & 0xffffu) << 16)); // tdim0 hi | tdim1 lo
  g1[3] = (int)(64u << 16);                     // tdim1 hi (0) | tile_dim0=64
  g1[4] = 32;                                   // tile_dim1=32, tile_dim2=0
  g1[5] = V;                                    // tensor_dim0_stride lo
  g1[6] = 0;                                    // stride0 hi | stride1 lo
  g1[7] = 0;
  v4i z4 = {0, 0, 0, 0};
  v8i z8 = {0, 0, 0, 0, 0, 0, 0, 0};
  __builtin_amdgcn_tensor_load_to_lds(g0, g1, z4, z4, z8, 0);
}

// D layout: lanes 0-15 col n0+lane rows m0+v; lanes 16-31 col n0+lane-16 rows m0+8+v.

// ---------------------------------------------------------------------------
// Kernel 1: attention scores + softmax + attn_applied. 1 block = 16 batch rows.
// ---------------------------------------------------------------------------
__global__ __launch_bounds__(128) void k_attention(
    const float* __restrict__ emb, const float* __restrict__ h0,
    const float* __restrict__ enc, const float* __restrict__ attn_W,
    const float* __restrict__ attn_b, float* __restrict__ out_attnw,
    float* __restrict__ ws_attn) {
  __shared__ float s_sc[16 * L];
  __shared__ float s_w[16 * L];
  const int lane = threadIdx.x & 31;
  const int wave = threadIdx.x >> 5;
  const int m0 = blockIdx.x * 16;

  {  // scores tile (16 rows x 16 cols) at n0 = wave*16; K over concat[emb|h0]
    const int n0 = wave * 16;
    v8f acc = {};
    for (int k0 = 0; k0 < H; k0 += 32) {
      v16h a = load_a_rm(emb, H, m0, k0, lane);
      v16h b = load_b_kn(attn_W, L, k0, n0, lane, L);
      acc = wmma_f16(a, b, acc);
    }
    for (int k0 = 0; k0 < H; k0 += 32) {
      v16h a = load_a_rm(h0, H, m0, k0, lane);
      v16h b = load_b_kn(attn_W, L, H + k0, n0, lane, L);
      acc = wmma_f16(a, b, acc);
    }
    const int half = lane >> 4;
    const int n = n0 + (lane & 15);
    const float bias = attn_b[n];
#pragma unroll
    for (int v = 0; v < 8; ++v) s_sc[(half * 8 + v) * L + n] = acc[v] + bias;
  }
  __syncthreads();

  if (threadIdx.x < 16) {  // per-row softmax over L=64
    const int r = threadIdx.x;
    float m = -3.4e38f;
    for (int c = 0; c < L; ++c) m = fmaxf(m, s_sc[r * L + c]);
    float s = 0.f;
    for (int c = 0; c < L; ++c) s += __expf(s_sc[r * L + c] - m);
    const float inv = 1.f / s;
    for (int c = 0; c < L; ++c) {
      const float w = __expf(s_sc[r * L + c] - m) * inv;
      s_w[r * L + c] = w;
      out_attnw[(m0 + r) * L + c] = w;
    }
  }
  __syncthreads();

  // attn_applied = s_w (16 x 64) @ enc (64 x 1024); wave covers 256 columns
#pragma unroll 4
  for (int t = 0; t < 16; ++t) {
    const int n0 = wave * 256 + t * 16;
    v8f acc = {};
    {
      v16h a = load_a_rm(s_w, L, 0, 0, lane);
      v16h b = load_b_kn(enc, H, 0, n0, lane, H);
      acc = wmma_f16(a, b, acc);
    }
    {
      v16h a = load_a_rm(s_w, L, 0, 32, lane);
      v16h b = load_b_kn(enc, H, 32, n0, lane, H);
      acc = wmma_f16(a, b, acc);
    }
    const int half = lane >> 4;
    const int n = n0 + (lane & 15);
#pragma unroll
    for (int v = 0; v < 8; ++v)
      ws_attn[(long)(m0 + half * 8 + v) * H + n] = acc[v];
  }
}

// ---------------------------------------------------------------------------
// Kernel 2: x = relu(concat[emb|attn_applied] @ comb_W + comb_b)
// ---------------------------------------------------------------------------
__global__ __launch_bounds__(128) void k_combine(
    const float* __restrict__ emb, const float* __restrict__ ws_attn,
    const float* __restrict__ comb_W, const float* __restrict__ comb_b,
    float* __restrict__ ws_x) {
  const int lane = threadIdx.x & 31;
  const int wave = threadIdx.x >> 5;
  const int m0 = blockIdx.x * 16;
  const int n0 = blockIdx.y * 64 + wave * 16;
  v8f acc = {};
  for (int k0 = 0; k0 < H; k0 += 32) {
    v16h a = load_a_rm(emb, H, m0, k0, lane);
    v16h b = load_b_kn(comb_W, H, k0, n0, lane, H);
    __builtin_prefetch(comb_W + (long)(k0 + 32) * H + n0 + lane, 0, 1);
    acc = wmma_f16(a, b, acc);
  }
  for (int k0 = 0; k0 < H; k0 += 32) {
    v16h a = load_a_rm(ws_attn, H, m0, k0, lane);
    v16h b = load_b_kn(comb_W, H, H + k0, n0, lane, H);
    acc = wmma_f16(a, b, acc);
  }
  const int half = lane >> 4;
  const int n = n0 + (lane & 15);
  const float bias = comb_b[n];
#pragma unroll
  for (int v = 0; v < 8; ++v)
    ws_x[(long)(m0 + half * 8 + v) * H + n] = fmaxf(acc[v] + bias, 0.f);
}

// ---------------------------------------------------------------------------
// Kernel 3: fused GRU cell. One wave per 16x16 (batch, hidden) tile holds all
// six gate accumulators (r,z,n x input/hidden) and gates in registers.
// ---------------------------------------------------------------------------
__global__ __launch_bounds__(128) void k_gru(
    const float* __restrict__ x, const float* __restrict__ h0,
    const float* __restrict__ W_ih, const float* __restrict__ W_hh,
    const float* __restrict__ b_ih, const float* __restrict__ b_hh,
    float* __restrict__ h1) {
  const int lane = threadIdx.x & 31;
  const int wave = threadIdx.x >> 5;
  const int tile = blockIdx.x * 4 + wave;      // (B/16)*(H/16) = 2048 tiles
  const int m0 = (tile >> 6) * 16;             // 64 j-tiles per batch tile
  const int j0 = (tile & 63) * 16;
  v8f ir = {}, iz = {}, in_ = {}, hr = {}, hz = {}, hn = {};
  for (int k0 = 0; k0 < H; k0 += 32) {
    v16h ax = load_a_rm(x, H, m0, k0, lane);
    v16h ah = load_a_rm(h0, H, m0, k0, lane);
    ir  = wmma_f16(ax, load_b_nk(W_ih, H, k0, j0,         lane), ir);
    iz  = wmma_f16(ax, load_b_nk(W_ih, H, k0, j0 + H,     lane), iz);
    in_ = wmma_f16(ax, load_b_nk(W_ih, H, k0, j0 + 2 * H, lane), in_);
    hr  = wmma_f16(ah, load_b_nk(W_hh, H, k0, j0,         lane), hr);
    hz  = wmma_f16(ah, load_b_nk(W_hh, H, k0, j0 + H,     lane), hz);
    hn  = wmma_f16(ah, load_b_nk(W_hh, H, k0, j0 + 2 * H, lane), hn);
  }
  const int half = lane >> 4;
  const int j = j0 + (lane & 15);
  const float bir = b_ih[j], biz = b_ih[j + H], bin = b_ih[j + 2 * H];
  const float bhr = b_hh[j], bhz = b_hh[j + H], bhn = b_hh[j + 2 * H];
#pragma unroll
  for (int v = 0; v < 8; ++v) {
    const int m = m0 + half * 8 + v;
    const float r = 1.f / (1.f + __expf(-(ir[v] + bir + hr[v] + bhr)));
    const float z = 1.f / (1.f + __expf(-(iz[v] + biz + hz[v] + bhz)));
    const float hp = h0[(long)m * H + j];
    const float nn = tanhf(in_[v] + bin + r * (hn[v] + bhn));
    h1[(long)m * H + j] = (1.f - z) * nn + z * hp;
  }
}

// ---------------------------------------------------------------------------
// Kernel 4: logits = h1 @ out_W + out_b  (M=512, N=50257, K=1024).
// Block = 64 rows x 64 cols. Weight tiles (32K x 64N, 8 KB) are streamed by
// the Tensor Data Mover into double-buffered LDS (padded rows), one DMA
// instruction per tile, overlapped with compute via TENSORcnt. Each wave
// reuses its B fragment across 4 M-tiles (4 WMMAs per fragment).
// ---------------------------------------------------------------------------
__global__ __launch_bounds__(128) void k_out(
    const float* __restrict__ h1, const float* __restrict__ out_W,
    const float* __restrict__ out_b, float* __restrict__ logits) {
  __shared__ float s_w[2][32 * 65];             // 65-dword padded rows
  const int lane = threadIdx.x & 31;
  const int wave = threadIdx.x >> 5;
  const int m_base = blockIdx.x * 64;
  const int n_blk = blockIdx.y * 64;
  const unsigned rem = (unsigned)(V - n_blk);   // OOB columns zero-fill

  if (wave == 0)
    tdm_load_tile(out_W + n_blk,
                  (unsigned)(unsigned long long)(void*)&s_w[0][0], rem);

  v8f acc0 = {}, acc1 = {}, acc2 = {}, acc3 = {};
  int buf = 0;
  const int NK = H / 32;
  for (int kt = 0; kt < NK; ++kt) {
    if (wave == 0) {
      if (kt + 1 < NK) {
        tdm_load_tile(out_W + (long)(kt + 1) * 32 * V + n_blk,
                      (unsigned)(unsigned long long)(void*)&s_w[buf ^ 1][0],
                      rem);
        __builtin_amdgcn_s_wait_tensorcnt(1);   // current tile landed
      } else {
        __builtin_amdgcn_s_wait_tensorcnt(0);   // last tile landed
      }
    }
    __syncthreads();
    const int k0 = kt * 32;
    v16h b = load_b_lds(&s_w[buf][0], wave * 16, lane);
    v16h a0 = load_a_rm(h1, H, m_base, k0, lane);
    acc0 = wmma_f16(a0, b, acc0);
    v16h a1 = load_a_rm(h1, H, m_base + 16, k0, lane);
    acc1 = wmma_f16(a1, b, acc1);
    v16h a2 = load_a_rm(h1, H, m_base + 32, k0, lane);
    acc2 = wmma_f16(a2, b, acc2);
    v16h a3 = load_a_rm(h1, H, m_base + 48, k0, lane);
    acc3 = wmma_f16(a3, b, acc3);
    __syncthreads();                            // done reading s_w[buf]
    buf ^= 1;
  }
  const int half = lane >> 4;
  const int n = n_blk + wave * 16 + (lane & 15);
  if (n < V) {
    const float bias = out_b[n];
#pragma unroll
    for (int v = 0; v < 8; ++v) {
      logits[(long)(m_base +      half * 8 + v) * V + n] = acc0[v] + bias;
      logits[(long)(m_base + 16 + half * 8 + v) * V + n] = acc1[v] + bias;
      logits[(long)(m_base + 32 + half * 8 + v) * V + n] = acc2[v] + bias;
      logits[(long)(m_base + 48 + half * 8 + v) * V + n] = acc3[v] + bias;
    }
  }
}

// ---------------------------------------------------------------------------
// Kernel 5: per-row log-sum-exp over V.  Kernel 6: subtract (log_softmax).
// ---------------------------------------------------------------------------
__global__ __launch_bounds__(256) void k_lse(const float* __restrict__ logits,
                                             float* __restrict__ lse) {
  __shared__ float red[256];
  const int row = blockIdx.x;
  const float* p = logits + (long)row * V;
  float m = -3.4e38f;
  for (int i = threadIdx.x; i < V; i += 256) m = fmaxf(m, p[i]);
  red[threadIdx.x] = m;
  __syncthreads();
  for (int s = 128; s > 0; s >>= 1) {
    if (threadIdx.x < s)
      red[threadIdx.x] = fmaxf(red[threadIdx.x], red[threadIdx.x + s]);
    __syncthreads();
  }
  m = red[0];
  __syncthreads();
  float sum = 0.f;
  for (int i = threadIdx.x; i < V; i += 256) sum += __expf(p[i] - m);
  red[threadIdx.x] = sum;
  __syncthreads();
  for (int s = 128; s > 0; s >>= 1) {
    if (threadIdx.x < s) red[threadIdx.x] += red[threadIdx.x + s];
    __syncthreads();
  }
  if (threadIdx.x == 0) lse[row] = m + __logf(red[0]);
}

__global__ __launch_bounds__(256) void k_sub(float* __restrict__ logits,
                                             const float* __restrict__ lse) {
  const int row = blockIdx.y;
  const int col = blockIdx.x * 256 + threadIdx.x;
  if (col < V) logits[(long)row * V + col] -= lse[row];
}

// ---------------------------------------------------------------------------

extern "C" void kernel_launch(void* const* d_in, const int* in_sizes, int n_in,
                              void* d_out, int out_size, void* d_ws,
                              size_t ws_size, hipStream_t stream) {
  const float* emb    = (const float*)d_in[0];
  const float* h0     = (const float*)d_in[1];
  const float* enc    = (const float*)d_in[2];
  const float* attn_W = (const float*)d_in[3];
  const float* attn_b = (const float*)d_in[4];
  const float* comb_W = (const float*)d_in[5];
  const float* comb_b = (const float*)d_in[6];
  const float* W_ih   = (const float*)d_in[7];
  const float* W_hh   = (const float*)d_in[8];
  const float* b_ih   = (const float*)d_in[9];
  const float* b_hh   = (const float*)d_in[10];
  const float* out_W  = (const float*)d_in[11];
  const float* out_b  = (const float*)d_in[12];

  // d_out = [log_softmax (B*V)] [h1 (B*H)] [attn_weights (B*L)]
  float* out_logits = (float*)d_out;
  float* out_h1     = out_logits + (long)B * V;
  float* out_attnw  = out_h1 + (long)B * H;

  float* ws      = (float*)d_ws;
  float* ws_attn = ws;                   // B*H floats
  float* ws_x    = ws + (long)B * H;     // B*H floats
  float* ws_lse  = ws_x + (long)B * H;   // B floats

  k_attention<<<dim3(B / 16), 128, 0, stream>>>(emb, h0, enc, attn_W, attn_b,
                                                out_attnw, ws_attn);
  k_combine<<<dim3(B / 16, H / 64), 128, 0, stream>>>(emb, ws_attn, comb_W,
                                                      comb_b, ws_x);
  k_gru<<<dim3((B / 16) * (H / 16) / 4), 128, 0, stream>>>(ws_x, h0, W_ih, W_hh,
                                                           b_ih, b_hh, out_h1);
  k_out<<<dim3(B / 64, (V + 63) / 64), 128, 0, stream>>>(out_h1, out_W, out_b,
                                                         out_logits);
  k_lse<<<dim3(B), 256, 0, stream>>>(out_logits, ws_lse);
  k_sub<<<dim3((V + 255) / 256, B), 256, 0, stream>>>(out_logits, ws_lse);
}
